// DozatLstmCell_39092792328849
// MI455X (gfx1250) — compile-verified, matrix operand order checked by reading, not run on hardware
//
#include <hip/hip_runtime.h>

#define BB 32
#define TT 1024
#define DD 512
#define HH 512
#define FH 2048   // 4*HH

typedef __attribute__((ext_vector_type(16))) __bf16 v16bf;
typedef __attribute__((ext_vector_type(8)))  float  v8f;

#if defined(__has_builtin)
#if __has_builtin(__builtin_amdgcn_tensor_load_to_lds) && __has_builtin(__builtin_amdgcn_s_wait_tensorcnt)
#define USE_TDM 1
#endif
#endif

#ifdef USE_TDM
typedef __attribute__((ext_vector_type(4))) unsigned tdm_u4;
typedef __attribute__((ext_vector_type(8))) int      tdm_i8;
typedef __attribute__((ext_vector_type(4))) int      tdm_i4;
#endif

struct alignas(16) U4 { unsigned x, y, z, w; };
struct U32B { U4 lo, hi; };

static __device__ __forceinline__ unsigned short f2bf(float f) {
  unsigned u = __builtin_bit_cast(unsigned, f);
  unsigned r = (u + 0x7FFFu + ((u >> 16) & 1u)) >> 16;  // round-to-nearest-even
  return (unsigned short)r;
}

// Load one 16x32 bf16 WMMA operand tile (ISA 7.12.2 striping):
// lanes 0-15 hold K=0..7 / 16..23 of row (lane&15); lanes 16-31 hold K=8..15 / 24..31.
// Two 16-byte loads per lane (global_load_b128 or ds_load_b128 by address space).
static __device__ __forceinline__ v16bf ld_tile(const unsigned short* rowBase, int k0, int half) {
  U32B t;
  t.lo = *(const U4*)(rowBase + k0 + half * 8);
  t.hi = *(const U4*)(rowBase + k0 + 16 + half * 8);
  return __builtin_bit_cast(v16bf, t);
}

static __device__ __forceinline__ v8f wmma_bf16(v16bf a, v16bf b, v8f c) {
  return __builtin_amdgcn_wmma_f32_16x16x32_bf16(
      /*neg_a=*/false, a, /*neg_b=*/false, b,
      /*c_mod=*/(short)0, c, /*reuse_a=*/false, /*reuse_b=*/false);
}

static __device__ __forceinline__ void grid_barrier(unsigned* sync, unsigned target) {
  __threadfence();
  __syncthreads();
  if (threadIdx.x == 0) {
    __hip_atomic_fetch_add(sync, 1u, __ATOMIC_ACQ_REL, __HIP_MEMORY_SCOPE_AGENT);
    while (__hip_atomic_load(sync, __ATOMIC_ACQUIRE, __HIP_MEMORY_SCOPE_AGENT) < target) {
      __builtin_amdgcn_s_sleep(8);
    }
  }
  __syncthreads();
}

// ---------------- fp32 -> bf16 conversion ----------------
__global__ __launch_bounds__(256) void cvt_bf16(const float* __restrict__ src,
                                                unsigned short* __restrict__ dst, int n) {
  int i = (blockIdx.x * 256 + threadIdx.x) * 4;
  if (i + 3 < n) {
#pragma unroll
    for (int u = 0; u < 4; ++u) dst[i + u] = f2bf(src[i + u]);
  }
}

// ---------------- px = x . W_in^T  (tbk layout) ----------------
// One wave computes a 32(M) x 64(N) block: 2 M-tiles x 4 N-tiles, software-pipelined
// double-buffered operand loads so WMMAs overlap the next iteration's global loads.
__global__ __launch_bounds__(256) void px_gemm(const unsigned short* __restrict__ xbf,
                                               const unsigned short* __restrict__ wbf,
                                               float* __restrict__ px) {
  const int gw     = (blockIdx.x * 256 + threadIdx.x) >> 5;  // global wave id
  const int nStrip = gw & 31;        // 32 strips of 64 cols over FH
  const int mTile  = gw >> 5;        // 1024 blocks of 32 rows over B*T
  const int lane = threadIdx.x & 31;
  const int half = lane >> 4;
  const int col  = lane & 15;

  const int rr0 = mTile * 32 + col;        // (t,b)-flattened rows; x is (B,T,D)
  const int rr1 = rr0 + 16;
  const unsigned short* rowA0 = xbf + ((size_t)(rr0 & 31) * TT + (rr0 >> 5)) * DD;
  const unsigned short* rowA1 = xbf + ((size_t)(rr1 & 31) * TT + (rr1 >> 5)) * DD;
  const int n0 = nStrip * 64;
  const unsigned short* rowB = wbf + (size_t)(n0 + col) * DD;

  const v8f zz = {0.f, 0.f, 0.f, 0.f, 0.f, 0.f, 0.f, 0.f};
  v8f a00 = zz, a01 = zz, a02 = zz, a03 = zz;
  v8f a10 = zz, a11 = zz, a12 = zz, a13 = zz;

  v16bf ta0 = ld_tile(rowA0, 0, half);
  v16bf ta1 = ld_tile(rowA1, 0, half);
  v16bf tb0 = ld_tile(rowB,           0, half);
  v16bf tb1 = ld_tile(rowB + 16 * DD, 0, half);
  v16bf tb2 = ld_tile(rowB + 32 * DD, 0, half);
  v16bf tb3 = ld_tile(rowB + 48 * DD, 0, half);

  for (int k0 = 0; k0 < DD - 32; k0 += 32) {
    const int kn = k0 + 32;
    v16bf na0 = ld_tile(rowA0, kn, half);
    v16bf na1 = ld_tile(rowA1, kn, half);
    v16bf nb0 = ld_tile(rowB,           kn, half);
    v16bf nb1 = ld_tile(rowB + 16 * DD, kn, half);
    v16bf nb2 = ld_tile(rowB + 32 * DD, kn, half);
    v16bf nb3 = ld_tile(rowB + 48 * DD, kn, half);
    a00 = wmma_bf16(ta0, tb0, a00);
    a01 = wmma_bf16(ta0, tb1, a01);
    a02 = wmma_bf16(ta0, tb2, a02);
    a03 = wmma_bf16(ta0, tb3, a03);
    a10 = wmma_bf16(ta1, tb0, a10);
    a11 = wmma_bf16(ta1, tb1, a11);
    a12 = wmma_bf16(ta1, tb2, a12);
    a13 = wmma_bf16(ta1, tb3, a13);
    ta0 = na0; ta1 = na1; tb0 = nb0; tb1 = nb1; tb2 = nb2; tb3 = nb3;
  }
  a00 = wmma_bf16(ta0, tb0, a00);
  a01 = wmma_bf16(ta0, tb1, a01);
  a02 = wmma_bf16(ta0, tb2, a02);
  a03 = wmma_bf16(ta0, tb3, a03);
  a10 = wmma_bf16(ta1, tb0, a10);
  a11 = wmma_bf16(ta1, tb1, a11);
  a12 = wmma_bf16(ta1, tb2, a12);
  a13 = wmma_bf16(ta1, tb3, a13);

#pragma unroll
  for (int r = 0; r < 8; ++r) {
    size_t o0 = (size_t)(mTile * 32 + r + 8 * half) * FH + n0 + col;
    px[o0]      = a00[r];
    px[o0 + 16] = a01[r];
    px[o0 + 32] = a02[r];
    px[o0 + 48] = a03[r];
    size_t o1 = o0 + (size_t)16 * FH;
    px[o1]      = a10[r];
    px[o1 + 16] = a11[r];
    px[o1 + 32] = a12[r];
    px[o1 + 48] = a13[r];
  }
}

// ---------------- persistent sequential LSTM scan ----------------
// 32 workgroups; WG g owns hidden slice j in [16g, 16g+16).
// Waves: mtile = wave&1 (batches 0-15 / 16-31), gate = wave>>1 (i,f,g,o).
// W_h slice (4 contiguous 16x512 bf16 tiles = 64KB) staged into dynamic LDS via the
// Tensor Data Mover (one 2-D descriptor per gate, issued by waves 0-3), then per-step
// B-operand tile loads are ds_load_b128. h crosses WGs via a global bf16 buffer.
__global__ __launch_bounds__(256) void lstm_scan(const float* __restrict__ px,
                                                 const unsigned short* __restrict__ whb,
                                                 const float* __restrict__ bias,
                                                 const long long* __restrict__ lengths,
                                                 unsigned short* __restrict__ hbf,
                                                 float* __restrict__ out,
                                                 unsigned* sync) {
  extern __shared__ unsigned short whLds[];   // 64 * 512 bf16 = 65536 bytes
  __shared__ float zS[4][BB][16];
  __shared__ float hS[BB][16];
  __shared__ float cS[BB][16];
  __shared__ long long lenS[BB];

  const int g    = blockIdx.x;
  const int j0   = g << 4;
  const int tid  = threadIdx.x;
  const int wave = tid >> 5;
  const int lane = tid & 31;
  const int half = lane >> 4;
  const int col  = lane & 15;
  const int mtile = wave & 1;
  const int gate  = wave >> 1;

#ifdef USE_TDM
  // TDM staging: gate q's slice = W_h rows [q*HH + j0, q*HH + j0 + 16), a contiguous
  // 16x512 bf16 tile. D# per ISA 8.3/8.4: 2-D tensor, data_size=2B,
  // tile_dim0=512 (contiguous K), tile_dim1=16 (rows), dim0 stride=512.
  if (wave < 4) {                      // uniform per-wave branch; TDM ignores EXEC
    const int q = wave;
    unsigned long long ga =
        (unsigned long long)(const void*)(whb + (size_t)(q * HH + j0) * HH);
    unsigned lds_byte = __builtin_amdgcn_groupstaticsize() + (unsigned)(q * 16 * HH * 2);
    tdm_u4 g0;
    g0[0] = 1u;                                            // count=1, user mode
    g0[1] = lds_byte;                                      // lds_addr (bytes)
    g0[2] = (unsigned)(ga & 0xFFFFFFFFu);                  // global_addr[31:0]
    g0[3] = (unsigned)((ga >> 32) & 0x01FFFFFFu) | (2u << 30);  // addr[56:32] | type=2
    tdm_i8 g1;
    g1[0] = (int)(1u << 16);                               // wg_mask=0, data_size=1 (2B)
    g1[1] = (int)((unsigned)(DD & 0xFFFF) << 16);          // tensor_dim0[15:0]=512
    g1[2] = (int)((DD >> 16) | (16u << 16));               // tensor_dim0[hi] | tensor_dim1=16
    g1[3] = (int)((unsigned)DD << 16);                     // tile_dim0=512
    g1[4] = 16;                                            // tile_dim1=16, tile_dim2=0
    g1[5] = DD;                                            // tensor_dim0_stride=512 (lo32)
    g1[6] = 0;                                             // stride hi | dim1_stride lo
    g1[7] = 0;                                             // dim1_stride hi
    tdm_i4 gz4 = {0, 0, 0, 0};                             // groups 2/3 unused (2-D)
    tdm_i8 gz8 = {0, 0, 0, 0, 0, 0, 0, 0};                 // extra group (6-arg form)
    __builtin_amdgcn_tensor_load_to_lds(g0, g1, gz4, gz4, gz8, 0);
  }
  __builtin_amdgcn_s_wait_tensorcnt(0);
#else
  for (int chunk = tid; chunk < (64 * HH) / 8; chunk += 256) {
    int e  = chunk * 8;
    int lr = e >> 9;
    int kk = e & (HH - 1);
    int q  = lr >> 4, c = lr & 15;
    *(U4*)(whLds + e) = *(const U4*)(whb + (size_t)(q * HH + j0 + c) * HH + kk);
  }
#endif

  // init state (h=c=0), including our slice of the global bf16 h
  for (int idx = tid; idx < BB * 16; idx += 256) {
    int m = idx >> 4, j = idx & 15;
    hS[m][j] = 0.f;
    cS[m][j] = 0.f;
    hbf[m * HH + j0 + j] = 0;
  }
  if (tid < BB) lenS[tid] = lengths[tid];

  unsigned phase = 1;
  grid_barrier(sync, 32u * phase);   // __syncthreads inside publishes LDS staging WG-wide
  ++phase;

  const unsigned short* rowA = hbf + (size_t)(mtile * 16 + col) * HH;   // h row (A, global)
  const unsigned short* rowB = whLds + (gate * 16 + col) * HH;          // W_h row (B, LDS)
  const int n = gate * HH + j0 + col;                                   // z column
  const float bn = bias[n];

  for (int t = 0; t < TT; ++t) {
    v8f acc = {0.f, 0.f, 0.f, 0.f, 0.f, 0.f, 0.f, 0.f};
#pragma unroll
    for (int k0 = 0; k0 < HH; k0 += 32) {
      v16bf a = ld_tile(rowA, k0, half);
      v16bf b = ld_tile(rowB, k0, half);
      acc = wmma_bf16(a, b, acc);
    }
    const float* pxt = px + (size_t)t * BB * FH;
#pragma unroll
    for (int r = 0; r < 8; ++r) {
      int m = mtile * 16 + r + 8 * half;
      zS[gate][m][col] = acc[r] + pxt[(size_t)m * FH + n] + bn;
    }
    __syncthreads();

    const float* pxn = px + (size_t)(t + 1) * BB * FH;  // prefetch next step's px
#pragma unroll
    for (int p = 0; p < 2; ++p) {
      int idx = tid + p * 256;
      int m = idx >> 4, j = idx & 15;
      float zi = zS[0][m][j], zf = zS[1][m][j], zg = zS[2][m][j], zo = zS[3][m][j];
      float ig = 1.f / (1.f + __expf(-zi));
      float fg = 1.f / (1.f + __expf(-zf));
      float gg = tanhf(zg);
      float og = 1.f / (1.f + __expf(-zo));
      float c_old = cS[m][j];
      float c_new = fmaf(ig, gg, fg * c_old);
      float h_new = og * tanhf(c_new);
      bool valid = ((long long)t < lenS[m]);
      float c2 = valid ? c_new : c_old;
      float h2 = valid ? h_new : hS[m][j];
      cS[m][j] = c2;
      hS[m][j] = h2;
      out[((size_t)m * TT + t) * HH + j0 + j] = valid ? h_new : 0.f;
      hbf[m * HH + j0 + j] = f2bf(h2);
      if (t + 1 < TT) {
#pragma unroll
        for (int q = 0; q < 4; ++q)
          __builtin_prefetch(pxn + (size_t)m * FH + q * HH + j0 + j, 0, 3);
      }
    }
    grid_barrier(sync, 32u * phase);   // all WGs published h for step t
    ++phase;
  }

  // finals: h_f then c_f appended after outputs
  const size_t BTH = (size_t)BB * TT * HH;
  for (int idx = tid; idx < BB * 16; idx += 256) {
    int m = idx >> 4, j = idx & 15;
    out[BTH + (size_t)m * HH + j0 + j]                   = hS[m][j];
    out[BTH + (size_t)BB * HH + (size_t)m * HH + j0 + j] = cS[m][j];
  }
}

extern "C" void kernel_launch(void* const* d_in, const int* in_sizes, int n_in,
                              void* d_out, int out_size, void* d_ws, size_t ws_size,
                              hipStream_t stream) {
  const float* x           = (const float*)d_in[0];
  const long long* lengths = (const long long*)d_in[1];
  const float* W_in        = (const float*)d_in[2];
  const float* W_h         = (const float*)d_in[3];
  const float* bias        = (const float*)d_in[4];
  float* out               = (float*)d_out;

  char* ws = (char*)d_ws;
  size_t off = 0;
  unsigned* sync       = (unsigned*)(ws + off);        off += 256;
  float* px            = (float*)(ws + off);           off += (size_t)TT * BB * FH * sizeof(float);
  unsigned short* xbf  = (unsigned short*)(ws + off);  off += (size_t)BB * TT * DD * 2;
  unsigned short* winb = (unsigned short*)(ws + off);  off += (size_t)FH * DD * 2;
  unsigned short* whb  = (unsigned short*)(ws + off);  off += (size_t)FH * HH * 2;
  unsigned short* hbf  = (unsigned short*)(ws + off);  off += (size_t)BB * HH * 2;
  (void)in_sizes; (void)n_in; (void)out_size; (void)ws_size;

  (void)hipMemsetAsync(sync, 0, 256, stream);

  const int nx = BB * TT * DD;
  cvt_bf16<<<nx / 1024, 256, 0, stream>>>(x, xbf, nx);
  const int nw = FH * DD;
  cvt_bf16<<<nw / 1024, 256, 0, stream>>>(W_in, winb, nw);
  cvt_bf16<<<nw / 1024, 256, 0, stream>>>(W_h, whb, nw);

  // (B*T/32) M-blocks * (FH/64) N-strips = 32768 waves -> 4096 blocks of 8 waves
  px_gemm<<<(BB * TT / 32) * (FH / 64) / 8, 256, 0, stream>>>(xbf, winb, px);

  // 64KB dynamic LDS for the per-WG W_h slice (gfx1250 allows 320KB/WG)
  lstm_scan<<<HH / 16, 256, 65536, stream>>>(px, whb, bias, lengths, hbf, out, sync);
}
